// MTRL_AttModel_51144470561076
// MI455X (gfx1250) — matrix operand order repaired and checked by reading, not compiled, not run on hardware
//
#include <hip/hip_runtime.h>
#include <math.h>

// Problem constants (reference: B=64, N=1024, DIN=HID=128)
#define BB   64
#define NN   1024
#define DD   128
#define HH   128
#define RTOT (BB * NN)            // 65536 rows
#define NEGF 9000000000000000.0f  // reference NEG

// Workspace layout (requires ~50.5 MB):
//   [0, 96KB)          : Wq/Wk/Wv converted to bf16
//   [128KB, 128KB+48MB): q,k,v activations in bf16  (3 * 65536 * 128 * 2B)
#define WS_QKV_OFF (1u << 17)

typedef __attribute__((ext_vector_type(16))) __bf16 v16bf;
typedef __attribute__((ext_vector_type(8)))  float  v8f;
typedef __attribute__((ext_vector_type(4)))  unsigned int v4u;
typedef __attribute__((ext_vector_type(4)))  int    v4i;
typedef __attribute__((ext_vector_type(8)))  int    v8i;

union BF16x16 { uint4 u[2]; v16bf f; };

// Load a 32-byte (16 x bf16) fragment from two 16-byte chunks.
__device__ __forceinline__ v16bf ldfrag2(const __bf16* p0, const __bf16* p1) {
    BF16x16 cv;
    cv.u[0] = *(const uint4*)p0;
    cv.u[1] = *(const uint4*)p1;
    return cv.f;
}

__device__ __forceinline__ v8f wmma_bf16(v16bf a, v16bf b, v8f c) {
    return __builtin_amdgcn_wmma_f32_16x16x32_bf16(
        /*neg_a=*/false, a, /*neg_b=*/false, b,
        /*c_mod=*/(short)0, c, /*reuse_a=*/false, /*reuse_b=*/false);
}

// Branchless tanh: 1 - 2/(exp(2x)+1).  exp->inf => +1, exp->0 => -1.
__device__ __forceinline__ float fast_tanh(float x) {
    float e = __expf(2.0f * x);
    return 1.0f - 2.0f * __builtin_amdgcn_rcpf(e + 1.0f);
}

// ---------------------------------------------------------------------------
// Kernel 0: convert weights f32 -> bf16 (order: Wq, Wk, Wv)
// ---------------------------------------------------------------------------
__global__ void MTRL_wconv_kernel(const float* __restrict__ Wq,
                                  const float* __restrict__ Wk,
                                  const float* __restrict__ Wv,
                                  __bf16* __restrict__ wbf) {
    int idx = blockIdx.x * 256 + threadIdx.x;      // 0 .. 3*128*128-1
    int p   = idx >> 14;                           // /16384
    int off = idx & 16383;
    const float* s = (p == 0) ? Wq : (p == 1) ? Wk : Wv;
    wbf[idx] = (__bf16)s[off];
}

// ---------------------------------------------------------------------------
// Kernel 1: q/k/v = tanh(x @ W^T + b), stored bf16.
// One wave owns a 16-row tile; all 3 projections, 8 col-tiles each.
// ---------------------------------------------------------------------------
__global__ __launch_bounds__(256) void MTRL_proj_kernel(
    const float*  __restrict__ x,
    const __bf16* __restrict__ wbf,
    const float*  __restrict__ bq,
    const float*  __restrict__ bk,
    const float*  __restrict__ bv,
    __bf16*       __restrict__ qkv) {
    __shared__ __bf16 wlds[128 * 136];             // padded: 34.8 KB

    const int lane = threadIdx.x & 31;
    const int wv   = threadIdx.x >> 5;
    const int hl   = lane >> 4;                    // lane half (K offset select)
    const int cc   = lane & 15;                    // row (A) / col (B,D) index
    const int i0   = (blockIdx.x * 8 + wv) * 16;   // first row of this wave's tile

    // --- A fragments: 16 rows x 128 K of x, converted to bf16 ---------------
    v16bf afrag[4];
    const float* xr = x + (size_t)(i0 + cc) * DD;
#pragma unroll
    for (int kg = 0; kg < 4; ++kg) {
        const float* pa = xr + 32 * kg + 8 * hl;        // K = 32kg + 8hl + [0,8)
        const float* pb = xr + 32 * kg + 16 + 8 * hl;   // K = 32kg + 16 + 8hl + [0,8)
        float4 a0 = *(const float4*)pa, a1 = *(const float4*)(pa + 4);
        float4 b0 = *(const float4*)pb, b1 = *(const float4*)(pb + 4);
        v16bf f;
        f[0]  = (__bf16)a0.x; f[1]  = (__bf16)a0.y; f[2]  = (__bf16)a0.z; f[3]  = (__bf16)a0.w;
        f[4]  = (__bf16)a1.x; f[5]  = (__bf16)a1.y; f[6]  = (__bf16)a1.z; f[7]  = (__bf16)a1.w;
        f[8]  = (__bf16)b0.x; f[9]  = (__bf16)b0.y; f[10] = (__bf16)b0.z; f[11] = (__bf16)b0.w;
        f[12] = (__bf16)b1.x; f[13] = (__bf16)b1.y; f[14] = (__bf16)b1.z; f[15] = (__bf16)b1.w;
        afrag[kg] = f;
    }

    for (int p = 0; p < 3; ++p) {
        const float*  bias = (p == 0) ? bq : (p == 1) ? bk : bv;
        const __bf16* Wp   = wbf + p * 16384;

        __syncthreads();
        {   // stage current W (128x128 bf16) into LDS, padded row stride 136
            int h   = threadIdx.x >> 1;
            int seg = (threadIdx.x & 1) * 64;
            const __bf16* src = Wp + h * 128 + seg;
            __bf16*       dst = wlds + h * 136 + seg;
#pragma unroll
            for (int e = 0; e < 8; ++e)
                *(uint4*)(dst + e * 8) = *(const uint4*)(src + e * 8);
        }
        __syncthreads();

        __bf16* outp = qkv + (size_t)p * RTOT * HH;
#pragma unroll
        for (int ct = 0; ct < 8; ++ct) {
            // Preload all four B fragments, then issue the WMMA chain.
            v16bf bfr[4];
#pragma unroll
            for (int kg = 0; kg < 4; ++kg) {
                const __bf16* wb = wlds + (16 * ct + cc) * 136 + 32 * kg + 8 * hl;
                bfr[kg] = ldfrag2(wb, wb + 16);
            }
            v8f acc = {0.f, 0.f, 0.f, 0.f, 0.f, 0.f, 0.f, 0.f};
#pragma unroll
            for (int kg = 0; kg < 4; ++kg)
                acc = wmma_bf16(afrag[kg], bfr[kg], acc);

            const int   hcol = 16 * ct + cc;
            const float bsc  = bias[hcol];
#pragma unroll
            for (int r = 0; r < 8; ++r) {
                float t = fast_tanh(acc[r] + bsc);
                outp[(size_t)(i0 + 8 * hl + r) * HH + hcol] = (__bf16)t;
            }
        }
    }
}

// ---------------------------------------------------------------------------
// Kernel 2: flash attention with mask.
// grid = B * (N/128); block = 256 (8 waves, 16 query rows each).
// Per 32-key step: K-tile staged via the Tensor Data Mover (TDM descriptor
// with LDS padding to the conflict-free stride), V^T staged manually,
// 8 WMMAs for S, online softmax, P transposed through per-wave LDS,
// 8 WMMAs for O += P V.
// ---------------------------------------------------------------------------
#define KST 136   // K-tile LDS row stride (elements): 64 DWORDs data + 4 DWORDs pad
#define VST 40    // V^T   LDS row stride
#define PST 40    // P scratch row stride

__global__ __launch_bounds__(256) void MTRL_attn_kernel(
    const __bf16* __restrict__ qkv,
    const int*    __restrict__ msk,
    float*        __restrict__ out) {
    __shared__ __bf16 kbuf[32 * KST];        // 8704 B (TDM destination)
    __shared__ __bf16 vtbuf[128 * VST];      // 10240 B  (V^T: [h][j])
    __shared__ __bf16 pscr[8][16 * PST];     // 10240 B  (per-wave P transpose)

    const int tid  = threadIdx.x;
    const int lane = tid & 31, wv = tid >> 5;
    const int hl   = lane >> 4, cc = lane & 15;
    const int b    = blockIdx.x >> 3;
    const int i0   = (blockIdx.x & 7) * 128 + wv * 16;   // query row base in batch

    const __bf16* Q = qkv;
    const __bf16* K = qkv + (size_t)RTOT * HH;
    const __bf16* V = qkv + (size_t)2 * RTOT * HH;

    // --- Q fragments (16 rows x 128 h), resident for the whole loop ---------
    v16bf qf[4];
    const __bf16* qr = Q + (size_t)(b * NN + i0 + cc) * HH;
#pragma unroll
    for (int kg = 0; kg < 4; ++kg)
        qf[kg] = ldfrag2(qr + 32 * kg + 8 * hl, qr + 32 * kg + 16 + 8 * hl);

    float rmax[8], rsum[8];
    v8f   o[8];
#pragma unroll
    for (int r = 0; r < 8; ++r) { rmax[r] = -INFINITY; rsum[r] = 0.f; }
    {
        v8f z = {0.f, 0.f, 0.f, 0.f, 0.f, 0.f, 0.f, 0.f};
#pragma unroll
        for (int ct = 0; ct < 8; ++ct) o[ct] = z;
    }

#if defined(__has_builtin) && __has_builtin(__builtin_amdgcn_tensor_load_to_lds)
    // --- TDM descriptor for the K tile (constant parts) ---------------------
    // Group0: count=1 | lds_addr | global_addr(57b) | type=2
    // Group1: data_size=2B, pad_enable, pad_interval=5 (64 DW = one 128-elem
    //         row), pad_amount=3 (4 DW = 8 elem) -> LDS row stride 136,
    //         tensor_dim0=128, tensor_dim1=32, tile 128x32, dim0_stride=128.
    const unsigned kbuf_lds = (unsigned)(size_t)(uintptr_t)&kbuf[0];
    const unsigned g1w0 = (1u << 16) | (1u << 20) | (5u << 22) | (3u << 25);
    v8i g1;
    g1[0] = (int)g1w0;
    g1[1] = (int)(128u << 16);   // tensor_dim0 = 128
    g1[2] = (int)(32u << 16);    // tensor_dim1 = 32
    g1[3] = (int)(128u << 16);   // tile_dim0   = 128
    g1[4] = 32;                  // tile_dim1   = 32
    g1[5] = 128;                 // tensor_dim0_stride = 128
    g1[6] = 0;
    g1[7] = 0;
    const v4i gz4 = {0, 0, 0, 0};
#if __clang_major__ >= 23
    const v8i gz8 = {0, 0, 0, 0, 0, 0, 0, 0};
#endif
#endif

    // staging partition for V^T: thread -> (h column, key half)
    const int vh  = tid & 127;           // h column
    const int vj0 = (tid >> 7) << 4;     // key group {0,16}

    for (int jb = 0; jb < NN; jb += 32) {
        __syncthreads();                 // previous tile's readers are done

        // --- stage K[jb..jb+32)[0..128) via Tensor Data Mover ---------------
#if defined(__has_builtin) && __has_builtin(__builtin_amdgcn_tensor_load_to_lds)
        if (wv == 0) {                   // TDM is per-wave; one wave issues it
            unsigned long long ga =
                (unsigned long long)(uintptr_t)(K + (size_t)(b * NN + jb) * HH);
            v4u g0;
            g0[0] = 1u;                                    // count = 1
            g0[1] = kbuf_lds;                              // lds_addr
            g0[2] = (unsigned)ga;                          // global_addr lo
            g0[3] = ((unsigned)(ga >> 32) & 0x01FFFFFFu) | (2u << 30);  // hi|type
#if __clang_major__ >= 23
            __builtin_amdgcn_tensor_load_to_lds(g0, g1, gz4, gz4, gz8, 0);
#else
            __builtin_amdgcn_tensor_load_to_lds(g0, g1, gz4, gz4, 0);
#endif
            __builtin_amdgcn_s_wait_tensorcnt(0);
        }
#else
        {   // fallback: manual staging
            const int jr = tid >> 3, ho = (tid & 7) << 4;
            const __bf16* ks = K + (size_t)(b * NN + jb + jr) * HH + ho;
            *(uint4*)&kbuf[jr * KST + ho]     = *(const uint4*)ks;
            *(uint4*)&kbuf[jr * KST + ho + 8] = *(const uint4*)(ks + 8);
        }
#endif
        // --- stage V^T: vtbuf[h][j] -----------------------------------------
        {
            const __bf16* vs = V + (size_t)(b * NN + jb + vj0) * HH + vh;
            __attribute__((aligned(16))) __bf16 tmp[16];
#pragma unroll
            for (int e = 0; e < 16; ++e) tmp[e] = vs[(size_t)e * HH];
            *(uint4*)&vtbuf[vh * VST + vj0]     = *(const uint4*)&tmp[0];
            *(uint4*)&vtbuf[vh * VST + vj0 + 8] = *(const uint4*)&tmp[8];
        }
        if (jb + 32 < NN)   // global_prefetch_b8 of next V tile
            __builtin_prefetch(V + (size_t)(b * NN + jb + 32 + vj0) * HH + vh, 0, 0);
        __syncthreads();

        // --- S = Q K^T : two 16x16 tiles, K-dim 128 -------------------------
        // Preload all 8 K fragments, then run the 8-WMMA chain back-to-back.
        v16bf kf[8];
#pragma unroll
        for (int kg = 0; kg < 4; ++kg) {
            const __bf16* k0 = &kbuf[cc * KST + 32 * kg + 8 * hl];
            const __bf16* k1 = &kbuf[(16 + cc) * KST + 32 * kg + 8 * hl];
            kf[kg]     = ldfrag2(k0, k0 + 16);
            kf[4 + kg] = ldfrag2(k1, k1 + 16);
        }
        v8f s0 = {0.f, 0.f, 0.f, 0.f, 0.f, 0.f, 0.f, 0.f};
        v8f s1 = s0;
#pragma unroll
        for (int kg = 0; kg < 4; ++kg) {
            s0 = wmma_bf16(qf[kg], kf[kg], s0);
            s1 = wmma_bf16(qf[kg], kf[4 + kg], s1);
        }

        // --- mask + online softmax ------------------------------------------
        const int* mrow = msk + (size_t)(b * NN + i0 + 8 * hl) * NN + jb;
        __bf16* pw = pscr[wv];
#pragma unroll
        for (int r = 0; r < 8; ++r) {
            int   m0 = mrow[(size_t)r * NN + cc];
            int   m1 = mrow[(size_t)r * NN + cc + 16];
            if (jb + 32 < NN)
                __builtin_prefetch(mrow + (size_t)r * NN + cc + 32, 0, 0);
            float a0 = m0 ? s0[r] : -NEGF;     // == s*m - NEG*(1-m) for m in {0,1}
            float a1 = m1 ? s1[r] : -NEGF;

            float lm = fmaxf(a0, a1);          // row max over 16 lanes of this half
#pragma unroll
            for (int off = 1; off < 16; off <<= 1)
                lm = fmaxf(lm, __shfl_xor(lm, off, 32));
            float nm = fmaxf(rmax[r], lm);
            float sc = __expf(rmax[r] - nm);
            rmax[r]  = nm;

            float p0 = __expf(a0 - nm);
            float p1 = __expf(a1 - nm);
            float ps = p0 + p1;
#pragma unroll
            for (int off = 1; off < 16; off <<= 1)
                ps += __shfl_xor(ps, off, 32);
            rsum[r] = rsum[r] * sc + ps;

#pragma unroll
            for (int ct = 0; ct < 8; ++ct) o[ct][r] *= sc;

            // P transpose through LDS (D layout -> A layout)
            pw[(8 * hl + r) * PST + cc]      = (__bf16)p0;
            pw[(8 * hl + r) * PST + cc + 16] = (__bf16)p1;
        }

        // --- O += P @ V  (K-dim = 32 keys, one WMMA per col-tile) -----------
        v16bf pf = ldfrag2(&pw[cc * PST + 16 * hl], &pw[cc * PST + 16 * hl + 8]);
#pragma unroll
        for (int g = 0; g < 8; g += 4) {    // preload 4 V frags, then 4 WMMAs
            v16bf vf[4];
#pragma unroll
            for (int t = 0; t < 4; ++t) {
                const __bf16* vp = &vtbuf[(16 * (g + t) + cc) * VST + 16 * hl];
                vf[t] = ldfrag2(vp, vp + 8);
            }
#pragma unroll
            for (int t = 0; t < 4; ++t)
                o[g + t] = wmma_bf16(pf, vf[t], o[g + t]);
        }
    }

    // --- epilogue: normalize and store f32 ---------------------------------
#pragma unroll
    for (int r = 0; r < 8; ++r) {
        float inv   = __builtin_amdgcn_rcpf(rsum[r]);
        float* orow = out + (size_t)(b * NN + i0 + 8 * hl + r) * HH;
#pragma unroll
        for (int ct = 0; ct < 8; ++ct)
            orow[16 * ct + cc] = o[ct][r] * inv;
    }
}

// ---------------------------------------------------------------------------
// Host launcher. Inputs: 0:x 1:mask 2:Wv 3:bv 4:Wk 5:bk 6:Wq 7:bq
// ---------------------------------------------------------------------------
extern "C" void kernel_launch(void* const* d_in, const int* in_sizes, int n_in,
                              void* d_out, int out_size, void* d_ws, size_t ws_size,
                              hipStream_t stream) {
    (void)in_sizes; (void)n_in; (void)out_size; (void)ws_size;
    const float* x    = (const float*)d_in[0];
    const int*   mask = (const int*)d_in[1];
    const float* Wv   = (const float*)d_in[2];
    const float* bv   = (const float*)d_in[3];
    const float* Wk   = (const float*)d_in[4];
    const float* bk   = (const float*)d_in[5];
    const float* Wq   = (const float*)d_in[6];
    const float* bq   = (const float*)d_in[7];

    __bf16* wbf = (__bf16*)d_ws;
    __bf16* qkv = (__bf16*)((char*)d_ws + WS_QKV_OFF);
    float*  outp = (float*)d_out;

    // 3*128*128 = 49152 elems -> 192 blocks
    MTRL_wconv_kernel<<<192, 256, 0, stream>>>(Wq, Wk, Wv, wbf);
    // 65536 rows / 16 per wave / 8 waves per block = 512 blocks
    MTRL_proj_kernel<<<512, 256, 0, stream>>>(x, wbf, bq, bk, bv, qkv);
    // 64 batches * 8 row-tiles = 512 blocks
    MTRL_attn_kernel<<<512, 256, 0, stream>>>(qkv, mask, outp);
}
